// EstimateAdj_23596550324898
// MI455X (gfx1250) — matrix operand order repaired and checked by reading, not compiled, not run on hardware
//
#include <hip/hip_runtime.h>

// ---------------------------------------------------------------------------
// EstimateAdj fused pipeline for MI455X (gfx1250, wave32, WMMA).
//   rep  = relu(features @ W1 + b1) @ W2 + b2         (bf16 WMMA, f32 accum)
//   pw[e]= relu(dot(rep[e0[e]], rep[e1[e]]))          (orig edges only)
//   plus index copies to the output buffer (as float).
// ---------------------------------------------------------------------------

typedef __attribute__((ext_vector_type(16))) __bf16        v16bf;
typedef __attribute__((ext_vector_type(16))) float         v16f;
typedef __attribute__((ext_vector_type(8)))  float         v8f;

// single f32 -> bf16 bits via native fptrunc (RNE on AMDGPU)
__device__ __forceinline__ unsigned short bfbits(float f) {
    return __builtin_bit_cast(unsigned short, (__bf16)f);
}

// ---------------------------------------------------------------------------
// Pack W1 (128x64, K-major rows) and W2 (64x64) into the WMMA B-operand
// layout for v_wmma_f32_16x16x32_bf16:
//   B tile is 32x16 (KxN). Lane L holds column N = L&15.
//   Lanes 0-15: slots 0-7 -> K=0..7, slots 8-15 -> K=16..23; lanes 16-31:
//   slots 0-7 -> K=8..15, slots 8-15 -> K=24..31 (2 bf16 per VGPR, low first).
// Tile order: tile = kt*4 + nt  (kt over K/32, nt over N/16).
// Storage: pack[(tile*32 + L)*16 + s]
// ---------------------------------------------------------------------------
__global__ void pack_weights_kernel(const float* __restrict__ W1,
                                    const float* __restrict__ W2,
                                    unsigned short* __restrict__ p1,
                                    unsigned short* __restrict__ p2) {
    int t = blockIdx.x * blockDim.x + threadIdx.x;
    if (t < 16 * 512) {                 // W1: 4 k-tiles x 4 n-tiles
        int tile = t >> 9, idx = t & 511;
        int L = idx >> 4, s = idx & 15;
        int kt = tile >> 2, nt = tile & 3;
        int K = 32 * kt + (L >> 4) * 8 + (s & 7) + (s >> 3) * 16;
        int N = 16 * nt + (L & 15);
        p1[t] = bfbits(W1[K * 64 + N]);
    }
    if (t < 8 * 512) {                  // W2: 2 k-tiles x 4 n-tiles
        int tile = t >> 9, idx = t & 511;
        int L = idx >> 4, s = idx & 15;
        int kt = tile >> 2, nt = tile & 3;
        int K = 32 * kt + (L >> 4) * 8 + (s & 7) + (s >> 3) * 16;
        int N = 16 * nt + (L & 15);
        p2[t] = bfbits(W2[K * 64 + N]);
    }
}

// ---------------------------------------------------------------------------
// Fused two-layer MLP. 256 threads = 8 wave32 waves; each wave computes a
// 16x64 tile of rep. Weights staged in LDS once per workgroup.
// A-operand layout (16x32 bf16): lane L holds row M=L&15;
//   lanes 0-15 : slots 0-7 -> K=0..7,  slots 8-15 -> K=16..23
//   lanes 16-31: slots 0-7 -> K=8..15, slots 8-15 -> K=24..31
// ---------------------------------------------------------------------------
#define WMMA_BF16 __builtin_amdgcn_wmma_f32_16x16x32_bf16

__global__ __launch_bounds__(256) void mlp_kernel(
    const float* __restrict__ feat,
    const float* __restrict__ b1,
    const float* __restrict__ b2,
    const unsigned short* __restrict__ p1,
    const unsigned short* __restrict__ p2,
    float* __restrict__ rep, int Nrows) {

    __shared__ __align__(32) unsigned short sB1[16 * 512];   // 16 KB
    __shared__ __align__(32) unsigned short sB2[8 * 512];    //  8 KB
    __shared__ __align__(16) unsigned short sH[8][16 * 64];  // 16 KB (h per wave)

    const int tid = threadIdx.x;

    // cooperative stage of packed weights into LDS (verbatim copy)
    {
        const uint4* g1 = (const uint4*)p1;  uint4* l1 = (uint4*)sB1;
        for (int i = tid; i < 1024; i += 256) l1[i] = g1[i];
        const uint4* g2 = (const uint4*)p2;  uint4* l2 = (uint4*)sB2;
        for (int i = tid; i < 512; i += 256) l2[i] = g2[i];
    }
    __syncthreads();

    const int wave = tid >> 5;
    const int lane = tid & 31;
    const int half = lane >> 4;          // 0 or 1
    const int mr   = lane & 15;          // row within tile (also N within n-tile)
    const int m0   = (blockIdx.x * 8 + wave) * 16;
    if (m0 >= Nrows) return;             // wave-uniform; no barriers follow

    const v16bf* bB1 = (const v16bf*)sB1;
    const v16bf* bB2 = (const v16bf*)sB2;

    // ---------------- layer 1: h = relu(feat @ W1 + b1) ----------------
    int row = m0 + mr; if (row >= Nrows) row = Nrows - 1;   // never triggers (N%16==0)
    const float* frow = feat + (size_t)row * 128;

    // Hoist the entire 128-float row load (this lane's slices) so all 16
    // global_load_b128 are in flight before any pack/WMMA consumes them.
    float4 F[16];
#pragma unroll
    for (int kt = 0; kt < 4; ++kt) {
        const float* fr = frow + kt * 32 + half * 8;
        F[kt * 4 + 0] = *(const float4*)(fr + 0);
        F[kt * 4 + 1] = *(const float4*)(fr + 4);
        F[kt * 4 + 2] = *(const float4*)(fr + 16);
        F[kt * 4 + 3] = *(const float4*)(fr + 20);
    }

    // pack to bf16 A operands via native packed conversion (RNE)
    v16bf A[4];
#pragma unroll
    for (int kt = 0; kt < 4; ++kt) {
        float4 f0 = F[kt * 4 + 0], f1 = F[kt * 4 + 1];
        float4 f2 = F[kt * 4 + 2], f3 = F[kt * 4 + 3];
        v16f vf = { f0.x, f0.y, f0.z, f0.w, f1.x, f1.y, f1.z, f1.w,
                    f2.x, f2.y, f2.z, f2.w, f3.x, f3.y, f3.z, f3.w };
        A[kt] = __builtin_convertvector(vf, v16bf);
    }

    v8f c0 = {}, c1 = {}, c2 = {}, c3 = {};
#pragma unroll
    for (int kt = 0; kt < 4; ++kt) {
        c0 = WMMA_BF16(false, A[kt], false, bB1[(kt * 4 + 0) * 32 + lane], (short)0, c0, false, false);
        c1 = WMMA_BF16(false, A[kt], false, bB1[(kt * 4 + 1) * 32 + lane], (short)0, c1, false, false);
        c2 = WMMA_BF16(false, A[kt], false, bB1[(kt * 4 + 2) * 32 + lane], (short)0, c2, false, false);
        c3 = WMMA_BF16(false, A[kt], false, bB1[(kt * 4 + 3) * 32 + lane], (short)0, c3, false, false);
    }

    // bias + relu; spill h tile (16x64 bf16) to LDS to re-layout C -> A.
    // C layout: VGPR r, lane L: M = r + 8*half, N = 16n + mr.
    {
        float bn0 = b1[ 0 + mr], bn1 = b1[16 + mr], bn2 = b1[32 + mr], bn3 = b1[48 + mr];
        unsigned short* hb = &sH[wave][0];
#pragma unroll
        for (int r = 0; r < 8; ++r) {
            unsigned short* hp = hb + (r + 8 * half) * 64 + mr;
            hp[ 0] = bfbits(fmaxf(c0[r] + bn0, 0.f));
            hp[16] = bfbits(fmaxf(c1[r] + bn1, 0.f));
            hp[32] = bfbits(fmaxf(c2[r] + bn2, 0.f));
            hp[48] = bfbits(fmaxf(c3[r] + bn3, 0.f));
        }
    }
    // same-wave DS ops are in-order; compiler inserts s_wait_dscnt before reads.

    // ---------------- layer 2: rep = h @ W2 + b2 ----------------
    v8f d0 = {}, d1 = {}, d2 = {}, d3 = {};
    const unsigned short* hrow = &sH[wave][mr * 64];
#pragma unroll
    for (int kt = 0; kt < 2; ++kt) {
        const unsigned short* pA = hrow + kt * 32 + 8 * half;
        uint4 lo = *(const uint4*)(pA);
        uint4 hi = *(const uint4*)(pA + 16);
        struct { uint4 a, b; } pr = { lo, hi };
        v16bf a2 = __builtin_bit_cast(v16bf, pr);
        d0 = WMMA_BF16(false, a2, false, bB2[(kt * 4 + 0) * 32 + lane], (short)0, d0, false, false);
        d1 = WMMA_BF16(false, a2, false, bB2[(kt * 4 + 1) * 32 + lane], (short)0, d1, false, false);
        d2 = WMMA_BF16(false, a2, false, bB2[(kt * 4 + 2) * 32 + lane], (short)0, d2, false, false);
        d3 = WMMA_BF16(false, a2, false, bB2[(kt * 4 + 3) * 32 + lane], (short)0, d3, false, false);
    }

    float bo0 = b2[ 0 + mr], bo1 = b2[16 + mr], bo2 = b2[32 + mr], bo3 = b2[48 + mr];
#pragma unroll
    for (int r = 0; r < 8; ++r) {
        int orow = m0 + r + 8 * half;
        if (orow < Nrows) {
            float* o = rep + (size_t)orow * 64 + mr;
            o[ 0] = d0[r] + bo0;
            o[16] = d1[r] + bo1;
            o[32] = d2[r] + bo2;
            o[48] = d3[r] + bo3;
        }
    }
}

// ---------------------------------------------------------------------------
// Edge scoring: pw[e] = relu(dot(rep[e0[e]], rep[e1[e]])). rep is 25.6 MB ->
// L2-resident on the 192 MB L2, so the gathers run at L2 bandwidth.
// ---------------------------------------------------------------------------
__global__ __launch_bounds__(256) void edge_dot_kernel(
    const float* __restrict__ rep,
    const int* __restrict__ e0, const int* __restrict__ e1,
    float* __restrict__ pw, int E) {
    int e = blockIdx.x * blockDim.x + threadIdx.x;
    if (e >= E) return;
    const float4* a = (const float4*)(rep + (size_t)e0[e] * 64);
    const float4* b = (const float4*)(rep + (size_t)e1[e] * 64);
    float s = 0.f;
#pragma unroll
    for (int k = 0; k < 16; ++k) {
        float4 x = a[k], y = b[k];
        s += x.x * y.x + x.y * y.y + x.z * y.z + x.w * y.w;
    }
    pw[e] = fmaxf(s, 0.f);
}

// total_edge_index = concat([edge, pred], axis=1) written as float, row-major (2, E+P)
__global__ __launch_bounds__(256) void write_total_kernel(
    const int* __restrict__ e0, const int* __restrict__ e1,
    const int* __restrict__ p0, const int* __restrict__ p1,
    float* __restrict__ tot, int E, int P) {
    int t = blockIdx.x * blockDim.x + threadIdx.x;
    int T = E + P;
    if (t >= T) return;
    tot[t]     = (float)((t < E) ? e0[t] : p0[t - E]);
    tot[T + t] = (float)((t < E) ? e1[t] : p1[t - E]);
}

__global__ __launch_bounds__(256) void copy_idx_kernel(
    const int* __restrict__ src, float* __restrict__ dst, int n) {
    int t = blockIdx.x * blockDim.x + threadIdx.x;
    if (t < n) dst[t] = (float)src[t];
}

// ---------------------------------------------------------------------------
extern "C" void kernel_launch(void* const* d_in, const int* in_sizes, int n_in,
                              void* d_out, int out_size, void* d_ws, size_t ws_size,
                              hipStream_t stream) {
    const float* feat = (const float*)d_in[0];        // (N, 128) f32
    const int*   ei   = (const int*)d_in[1];          // (2, E)
    const int*   pei  = (const int*)d_in[2];          // (2, P)
    const float* W1   = (const float*)d_in[3];        // (128, 64)
    const float* b1   = (const float*)d_in[4];        // (64,)
    const float* W2   = (const float*)d_in[5];        // (64, 64)
    const float* b2   = (const float*)d_in[6];        // (64,)

    const int Nrows = in_sizes[0] / 128;              // 100000
    const int E     = in_sizes[1] / 2;                // 1,600,000
    const int P     = in_sizes[2] / 2;                //   800,000

    // workspace: packed bf16 weights (16 KB + 8 KB)
    unsigned short* p1 = (unsigned short*)d_ws;
    unsigned short* p2 = p1 + 16 * 512;

    // output layout (flat, reference return order, as float)
    float* rep = (float*)d_out;                       // N*64
    float* pw  = rep + (size_t)Nrows * 64;            // E
    float* tot = pw + E;                              // 2*(E+P)
    float* eif = tot + 2 * (size_t)(E + P);           // 2*E

    pack_weights_kernel<<<32, 256, 0, stream>>>(W1, W2, p1, p2);

    int mlp_blocks = (Nrows + 127) / 128;             // 8 waves x 16 rows per WG
    mlp_kernel<<<mlp_blocks, 256, 0, stream>>>(feat, b1, b2, p1, p2, rep, Nrows);

    edge_dot_kernel<<<(E + 255) / 256, 256, 0, stream>>>(rep, ei, ei + E, pw, E);

    write_total_kernel<<<((E + P) + 255) / 256, 256, 0, stream>>>(
        ei, ei + E, pei, pei + P, tot, E, P);

    copy_idx_kernel<<<(2 * E + 255) / 256, 256, 0, stream>>>(ei, eif, 2 * E);
}
